// CosineAttention_704374636756
// MI455X (gfx1250) — matrix operand order repaired and checked by reading, not compiled, not run on hardware
//
#include <hip/hip_runtime.h>
#include <hip/hip_bf16.h>

// Cosine attention (no softmax) reassociated into a pure GEMM chain:
//   out = (Q/||q||) @ [ (K/||k||)^T @ V ] , folded with W_out per head.
// All GEMMs use v_wmma_f32_16x16x32_bf16 (wave32, f32 accumulate).
// All WMMA B operands are pre-packed fragment-contiguous (32B/lane ->
// 2x global_load_b128); the K^T A operand is produced transposed so its
// loads are contiguous b128 as well.

namespace {

constexpr int kDim     = 512;
constexpr int kHeads   = 8;
constexpr int kDimHead = 64;
constexpr int kBatch   = 4;
constexpr int kSeqQ    = 1024;
constexpr int kSeqK    = 2048;
constexpr int kRowsQ   = kBatch * kSeqQ;   // 4096
constexpr int kRowsK   = kBatch * kSeqK;   // 8192

typedef __attribute__((ext_vector_type(16))) __bf16 v16bf;
typedef __attribute__((ext_vector_type(8)))  float  v8f;

__device__ __forceinline__ __bf16 f2bf(float f) {
  unsigned u = __float_as_uint(f);
  u += 0x7FFFu + ((u >> 16) & 1u);          // round-to-nearest-even
  unsigned short h = (unsigned short)(u >> 16);
  __bf16 r;
  __builtin_memcpy(&r, &h, sizeof(r));
  return r;
}

__device__ __forceinline__ v8f wmma_bf16(v16bf a, v16bf b, v8f c) {
  // (neg_a, A, neg_b, B, c_mod, C, reuse_a, reuse_b)
  return __builtin_amdgcn_wmma_f32_16x16x32_bf16(false, a, false, b,
                                                 (short)0, c, false, false);
}

// ---- WMMA fragment loaders (wave32, contiguous k = 16*(lane>>4) + e in A and B) ----------

// A 16x32 bf16 from row-major [rows x ld]: lane<16 -> row m0+lane,
// K = {k0..k0+15}; lane>=16 -> same row, K shifted by +16.
// Contiguous: two 16B loads per lane (global_load_b128).
__device__ __forceinline__ v16bf load_a_rm(const __bf16* __restrict__ src, int ld,
                                           int m0, int k0, int lane) {
  const int r  = lane & 15;
  const int kb = k0 + ((lane >> 4) << 4);
  const __bf16* p = src + (size_t)(m0 + r) * ld + kb;
  v16bf a;
#pragma unroll
  for (int i = 0; i < 8; ++i) a[i] = p[i];
#pragma unroll
  for (int i = 0; i < 8; ++i) a[8 + i] = p[8 + i];
  return a;
}

// Same but source is f32 (converted on load).
__device__ __forceinline__ v16bf load_a_f32(const float* __restrict__ src, int ld,
                                            int m0, int k0, int lane) {
  const int r  = lane & 15;
  const int kb = k0 + ((lane >> 4) << 4);
  const float* p = src + (size_t)(m0 + r) * ld + kb;
  v16bf a;
#pragma unroll
  for (int i = 0; i < 8; ++i) a[i] = f2bf(p[i]);
#pragma unroll
  for (int i = 0; i < 8; ++i) a[8 + i] = f2bf(p[8 + i]);
  return a;
}

// B 32x16 bf16 from PACKED layout: tile (kt,nt) stored as 32 lanes x 16 bf16
// contiguous; lane's fragment = one contiguous 32B chunk (2x b128 loads).
// Element e of lane L is B[kt*32 + (L>>4)*16 + e][nt*16 + (L&15)].
__device__ __forceinline__ v16bf load_b_pk(const __bf16* __restrict__ base,
                                           int ntiles, int kt, int nt, int lane) {
  const __bf16* p = base + (((size_t)kt * ntiles + nt) * 32 + lane) * 16;
  v16bf b;
#pragma unroll
  for (int e = 0; e < 16; ++e) b[e] = p[e];
  return b;
}

// ---- Kernel 0: pack Wq = W_qkv[:, :512] and W_out into B-fragment layout ---
__global__ void k_pack_weights(const float* __restrict__ W_qkv,
                               const float* __restrict__ W_out,
                               __bf16* __restrict__ Wq_pk,
                               __bf16* __restrict__ Wo_pk) {
  // One thread per 16-element fragment chunk: g = (kt*32 + nt)*32 + lane.
  const int g = blockIdx.x * blockDim.x + threadIdx.x;
  if (g >= (kDim / 32) * (kDim / 16) * 32) return;
  const int lane = g & 31;
  const int tile = g >> 5;
  const int kt = tile >> 5;            // tile = kt*32 + nt, nt in [0,32)
  const int nt = tile & 31;
  const int n  = nt * 16 + (lane & 15);
  const int k0 = kt * 32 + ((lane >> 4) << 4);
  __bf16* pq = Wq_pk + (size_t)g * 16;
  __bf16* po = Wo_pk + (size_t)g * 16;
#pragma unroll
  for (int e = 0; e < 16; ++e) {
    pq[e] = f2bf(W_qkv[(size_t)(k0 + e) * (3 * kDim) + n]);
    po[e] = f2bf(W_out[(size_t)(k0 + e) * kDim + n]);
  }
}

// ---- Kernel 1: LayerNorm -> bf16 (one wave per 512-wide row) ---------------
__global__ void k_layernorm_bf16(const float* __restrict__ x,
                                 const float* __restrict__ gamma,
                                 const float* __restrict__ beta,
                                 __bf16* __restrict__ y, int rows) {
  const int lane = threadIdx.x & 31;
  const int wave = threadIdx.x >> 5;
  const int row  = blockIdx.x * (blockDim.x >> 5) + wave;
  if (row >= rows) return;

  const float* xr = x + (size_t)row * kDim + lane * 16;
  float v[16];
#pragma unroll
  for (int i = 0; i < 16; ++i) v[i] = xr[i];

  float s = 0.f, sq = 0.f;
#pragma unroll
  for (int i = 0; i < 16; ++i) { s += v[i]; sq += v[i] * v[i]; }
#pragma unroll
  for (int m = 1; m < 32; m <<= 1) { s += __shfl_xor(s, m, 32); sq += __shfl_xor(sq, m, 32); }

  const float mu  = s * (1.0f / kDim);
  const float var = sq * (1.0f / kDim) - mu * mu;
  const float rs  = rsqrtf(var + 1e-5f);

  const float* g = gamma + lane * 16;
  const float* b = beta  + lane * 16;
  __bf16* yr = y + (size_t)row * kDim + lane * 16;
#pragma unroll
  for (int i = 0; i < 16; ++i) yr[i] = f2bf((v[i] - mu) * rs * g[i] + b[i]);
}

// ---- Kernel 2: projection GEMM + fused per-head L2 normalization -----------
// Y = X_ln[rows x 512] @ Wq[512 x 512], one 64-col head per block.y.
// Block = 128 threads = 4 waves; wave owns a 16x64 strip (4 WMMA accumulators).
// MODE 0: normalize rows, store row-major (Q-hat, A operand of final GEMM)
// MODE 1: normalize rows, store transposed per (b,h) (K-hat^T, A of KtV GEMM)
// MODE 2: no norm, store packed-B per (b,h)       (V, B operand of KtV GEMM)
template <int MODE>
__global__ void k_proj_norm(const __bf16* __restrict__ X,
                            const __bf16* __restrict__ Wq,
                            __bf16* __restrict__ Y) {
  const int lane = threadIdx.x & 31;
  const int wave = threadIdx.x >> 5;
  const int m0   = blockIdx.x * 64 + wave * 16;   // global row base
  const int h    = blockIdx.y;
  const int n0   = h * kDimHead;                  // head column base

  v8f acc[4] = {};
#pragma unroll 2
  for (int k0 = 0; k0 < kDim; k0 += 32) {
    v16bf a = load_a_rm(X, kDim, m0, k0, lane);
#pragma unroll
    for (int t = 0; t < 4; ++t) {
      v16bf b = load_b_pk(Wq, kDim / 16, k0 >> 5, (n0 >> 4) + t, lane);
      acc[t] = wmma_bf16(a, b, acc[t]);
    }
  }

  if (MODE != 2) {
    // Per-row L2 norm: reduce across the 16 lanes of each half-wave.
#pragma unroll
    for (int i = 0; i < 8; ++i) {
      float ss = 0.f;
#pragma unroll
      for (int t = 0; t < 4; ++t) { float xv = acc[t][i]; ss += xv * xv; }
      ss += __shfl_xor(ss, 1, 32);
      ss += __shfl_xor(ss, 2, 32);
      ss += __shfl_xor(ss, 4, 32);
      ss += __shfl_xor(ss, 8, 32);
      const float rsn = rsqrtf(ss + 1e-16f);
#pragma unroll
      for (int t = 0; t < 4; ++t) acc[t][i] *= rsn;
    }
  }

  const int half = lane >> 4;
  const int nl   = lane & 15;

  if (MODE == 0) {
    // Row-major [rows x 512]
#pragma unroll
    for (int t = 0; t < 4; ++t)
#pragma unroll
      for (int i = 0; i < 8; ++i)
        Y[(size_t)(m0 + half * 8 + i) * kDim + n0 + t * 16 + nl] = f2bf(acc[t][i]);
  } else if (MODE == 1) {
    // Transposed per (b,h): KnT[b][h][dimhead=64][seq=2048]; contiguous in seq.
    const int b   = m0 / kSeqK;
    const int kk0 = m0 % kSeqK;
    __bf16* dst = Y + ((size_t)(b * kHeads + h) * kDimHead) * kSeqK;
#pragma unroll
    for (int t = 0; t < 4; ++t) {
      __bf16* p = dst + (size_t)(t * 16 + nl) * kSeqK + kk0 + half * 8;
#pragma unroll
      for (int i = 0; i < 8; ++i) p[i] = f2bf(acc[t][i]);   // 16B contiguous
    }
  } else {
    // Packed-B per (b,h): [kt][nt=4][lane][16]; contiguous per lane.
    const int b   = m0 / kSeqK;
    const int kk0 = m0 % kSeqK;
    __bf16* dst = Y + (size_t)(b * kHeads + h) * kSeqK * kDimHead;
    const int laneSel = (((kk0 >> 4) & 1) << 4) | nl;
    const int kt = kk0 >> 5;
#pragma unroll
    for (int t = 0; t < 4; ++t) {
      __bf16* p = dst + (((size_t)kt * 4 + t) * 32 + laneSel) * 16 + half * 8;
#pragma unroll
      for (int i = 0; i < 8; ++i) p[i] = f2bf(acc[t][i]);   // 16B contiguous
    }
  }
}

// ---- Kernel 3: M_{b,h} = K-hat^T V  (64x64x2048, split-K + f32 atomics) ----
__global__ void k_kv_reduce(const __bf16* __restrict__ KnT,
                            const __bf16* __restrict__ Vpk,
                            float* __restrict__ M) {
  const int lane  = threadIdx.x & 31;
  const int wave  = threadIdx.x >> 5;
  const int bh    = blockIdx.x;                  // b*8 + h
  const int kbase = blockIdx.y * (kSeqK / 8);    // split-K chunk (256 rows)
  const int m0    = wave * 16;

  const __bf16* asrc = KnT + (size_t)bh * kDimHead * kSeqK;   // [64][2048]
  const __bf16* bsrc = Vpk + (size_t)bh * kSeqK * kDimHead;   // packed-B

  v8f acc[4] = {};
#pragma unroll 2
  for (int kk = 0; kk < kSeqK / 8; kk += 32) {
    const int k0 = kbase + kk;
    v16bf a = load_a_rm(asrc, kSeqK, m0, k0, lane);   // A = K-hat^T tile
#pragma unroll
    for (int t = 0; t < 4; ++t) {
      v16bf bb = load_b_pk(bsrc, 4, k0 >> 5, t, lane);
      acc[t] = wmma_bf16(a, bb, acc[t]);
    }
  }

  float* Mbh = M + (size_t)bh * kDimHead * kDimHead;
  const int half = lane >> 4, nl = lane & 15;
#pragma unroll
  for (int t = 0; t < 4; ++t)
#pragma unroll
    for (int i = 0; i < 8; ++i)
      atomicAdd(&Mbh[(m0 + half * 8 + i) * kDimHead + t * 16 + nl], acc[t][i]);
}

// ---- Kernel 4: P_b[h*64.., :] = M_{b,h} @ W_out[h*64.., :] -> packed bf16 --
__global__ void k_p_gemm(const float* __restrict__ M,
                         const __bf16* __restrict__ Wo_pk,
                         __bf16* __restrict__ Ppk) {
  const int lane = threadIdx.x & 31;
  const int wave = threadIdx.x >> 5;
  const int n0   = blockIdx.x * 64;
  const int h    = blockIdx.y;
  const int b    = blockIdx.z;
  const int m0   = wave * 16;

  const float* Mbh = M + ((size_t)(b * kHeads + h)) * kDimHead * kDimHead;

  v8f acc[4] = {};
#pragma unroll
  for (int k0 = 0; k0 < kDimHead; k0 += 32) {
    v16bf a = load_a_f32(Mbh, kDimHead, m0, k0, lane);
#pragma unroll
    for (int t = 0; t < 4; ++t) {
      v16bf bb = load_b_pk(Wo_pk, kDim / 16, (h * kDimHead + k0) >> 5,
                           (n0 >> 4) + t, lane);
      acc[t] = wmma_bf16(a, bb, acc[t]);
    }
  }

  // Store P directly in packed-B layout for the final GEMM.
  __bf16* Pb = Ppk + (size_t)b * kDim * kDim;
  const int half = lane >> 4, nl = lane & 15;
  const int kbase = h * kDimHead + m0;           // final-GEMM K index base
  const int laneSel = (((kbase >> 4) & 1) << 4) | nl;
  const int kt = kbase >> 5;
#pragma unroll
  for (int t = 0; t < 4; ++t) {
    const int nt = (n0 >> 4) + t;
    __bf16* p = Pb + (((size_t)kt * (kDim / 16) + nt) * 32 + laneSel) * 16 + half * 8;
#pragma unroll
    for (int i = 0; i < 8; ++i) p[i] = f2bf(acc[t][i]);     // 16B contiguous
  }
}

// ---- Kernel 5: out_b = Q-hat_b @ P_b + b_out  (1024x512x512, f32 out) ------
__global__ void k_final_gemm(const __bf16* __restrict__ Qn,
                             const __bf16* __restrict__ Ppk,
                             const float* __restrict__ bias,
                             float* __restrict__ out) {
  const int lane = threadIdx.x & 31;
  const int wave = threadIdx.x >> 5;
  const int b    = blockIdx.z;
  const int row0 = b * kSeqQ + blockIdx.x * 64 + wave * 16;
  const int n0   = blockIdx.y * 64;

  const __bf16* Pb = Ppk + (size_t)b * kDim * kDim;

  v8f acc[4] = {};
#pragma unroll 2
  for (int k0 = 0; k0 < kDim; k0 += 32) {
    __builtin_prefetch(Qn + (size_t)row0 * kDim + k0 + 64, 0, 1);
    v16bf a = load_a_rm(Qn, kDim, row0, k0, lane);
#pragma unroll
    for (int t = 0; t < 4; ++t) {
      v16bf bb = load_b_pk(Pb, kDim / 16, k0 >> 5, (n0 >> 4) + t, lane);
      acc[t] = wmma_bf16(a, bb, acc[t]);
    }
  }

  const int half = lane >> 4, nl = lane & 15;
#pragma unroll
  for (int t = 0; t < 4; ++t) {
    const int n = n0 + t * 16 + nl;
    const float bo = bias[n];
#pragma unroll
    for (int i = 0; i < 8; ++i)
      out[(size_t)(row0 + half * 8 + i) * kDim + n] = acc[t][i] + bo;
  }
}

} // namespace

extern "C" void kernel_launch(void* const* d_in, const int* in_sizes, int n_in,
                              void* d_out, int out_size, void* d_ws, size_t ws_size,
                              hipStream_t stream) {
  const float* q       = (const float*)d_in[0];
  const float* k       = (const float*)d_in[1];
  const float* v       = (const float*)d_in[2];
  const float* gamma   = (const float*)d_in[3];
  const float* beta    = (const float*)d_in[4];
  const float* W_qkv   = (const float*)d_in[5];
  const float* W_out   = (const float*)d_in[6];
  const float* b_out   = (const float*)d_in[7];
  float* out           = (float*)d_out;

  // Workspace carve-up (256B aligned), ~46 MB total.
  char* ws = (char*)d_ws;
  size_t off = 0;
  auto carve = [&](size_t bytes) -> void* {
    void* p = ws + off;
    off = (off + bytes + 255) & ~(size_t)255;
    return p;
  };
  __bf16* Wq_pk = (__bf16*)carve((size_t)kDim * kDim * 2);
  __bf16* Wo_pk = (__bf16*)carve((size_t)kDim * kDim * 2);
  __bf16* lnq   = (__bf16*)carve((size_t)kRowsQ * kDim * 2);
  __bf16* lnk   = (__bf16*)carve((size_t)kRowsK * kDim * 2);
  __bf16* lnv   = (__bf16*)carve((size_t)kRowsK * kDim * 2);
  __bf16* Qn    = (__bf16*)carve((size_t)kRowsQ * kDim * 2);
  __bf16* KnT   = (__bf16*)carve((size_t)kRowsK * kDim * 2);
  __bf16* Vpk   = (__bf16*)carve((size_t)kRowsK * kDim * 2);
  float*  Mws   = (float*) carve((size_t)kBatch * kHeads * kDimHead * kDimHead * 4);
  __bf16* Ppk   = (__bf16*)carve((size_t)kBatch * kDim * kDim * 2);

  // 0) weights -> bf16, packed into WMMA B-fragment layout
  k_pack_weights<<<(16384 + 255) / 256, 256, 0, stream>>>(W_qkv, W_out,
                                                          Wq_pk, Wo_pk);
  // 1) LayerNorm (one wave/row, 8 rows/block)
  k_layernorm_bf16<<<kRowsQ / 8, 256, 0, stream>>>(q, gamma, beta, lnq, kRowsQ);
  k_layernorm_bf16<<<kRowsK / 8, 256, 0, stream>>>(k, gamma, beta, lnk, kRowsK);
  k_layernorm_bf16<<<kRowsK / 8, 256, 0, stream>>>(v, gamma, beta, lnv, kRowsK);

  // 2) projections (+ fused per-head normalization; layout chosen per consumer)
  k_proj_norm<0><<<dim3(kRowsQ / 64, kHeads), 128, 0, stream>>>(lnq, Wq_pk, Qn);
  k_proj_norm<1><<<dim3(kRowsK / 64, kHeads), 128, 0, stream>>>(lnk, Wq_pk, KnT);
  k_proj_norm<2><<<dim3(kRowsK / 64, kHeads), 128, 0, stream>>>(lnv, Wq_pk, Vpk);

  // 3) M = K-hat^T V per (b,h); split-K (8 chunks) accumulated with f32 atomics
  hipMemsetAsync(Mws, 0, (size_t)kBatch * kHeads * kDimHead * kDimHead * 4, stream);
  k_kv_reduce<<<dim3(kBatch * kHeads, 8), 128, 0, stream>>>(KnT, Vpk, Mws);

  // 4) fold output projection: P_b = blockdiag(M) @ W_out  (packed store)
  k_p_gemm<<<dim3(kDim / 64, kHeads, kBatch), 128, 0, stream>>>(Mws, Wo_pk, Ppk);

  // 5) out = Q-hat @ P + b_out
  k_final_gemm<<<dim3(kSeqQ / 64, kDim / 64, kBatch), 128, 0, stream>>>(
      Qn, Ppk, b_out, out);
}
